// FancyMultiModalWindowAttentionBlock_37572373905980
// MI455X (gfx1250) — compile-verified, matrix operand order checked by reading, not run on hardware
//
#include <hip/hip_runtime.h>
#include <hip/hip_bf16.h>

typedef __attribute__((ext_vector_type(16))) _Float16 v16h;
typedef __attribute__((ext_vector_type(8)))  _Float16 v8h;
typedef __attribute__((ext_vector_type(8)))  float    v8f;

#define WMMA_F32_F16(a,b,c) __builtin_amdgcn_wmma_f32_16x16x32_f16(false,(a),false,(b),(short)0,(c),false,false)

#define Bv   2
#define Tv   32
#define Lv   64
#define Dv   1024
#define Hv   16
#define WSv  8
#define SHv  4
#define HDv  64
#define Sv   512      // WS*L
#define ROWS 4096     // B*T*L
#define D3   3072
#define D2   2048

static __device__ inline v8h ld8(const _Float16* p) { return *(const v8h*)p; }

static __device__ inline v16h mkfrag(v8h lo, v8h hi) {
  v16h r;
#pragma unroll
  for (int i = 0; i < 8; i++) { r[i] = lo[i]; r[i + 8] = hi[i]; }
  return r;
}

// window-order row -> source row of [B,T,L] with roll(-SH) folded in
static __device__ inline int src_row(int r) {
  int w = r >> 9;          // window (0..7): b*4 + widx
  int s = r & 511;         // token in window
  int b = w >> 2;
  int widx = w & 3;
  int tq = s >> 6;
  int l = s & 63;
  int tp = widx * WSv + tq;
  int tsrc = (tp + SHv) & 31;
  return ((b * Tv + tsrc) << 6) + l;
}

// ---------------------------------------------------------------- converts
__global__ void cvt_kernel(const float* __restrict__ s, _Float16* __restrict__ d, int n) {
  int i = blockIdx.x * 256 + threadIdx.x;
  if (i < n) d[i] = (_Float16)s[i];
}

__global__ void cvtT_kernel(const float* __restrict__ s, _Float16* __restrict__ d, int R, int C) {
  int i = blockIdx.x * 256 + threadIdx.x;
  if (i < R * C) {
    int r = i / C, c = i - r * C;
    d[(size_t)c * R + r] = (_Float16)s[i];   // dst is [C rows][R cols], k-contiguous
  }
}

// ---------------------------------------------------------------- QKV GEMM
// grid (32, 16, 3): 128-row x 64-col tiles per block; z selects Q/K/V.
// block = 128 (4 waves); each wave: 32 rows x 64 cols, 8 WMMAs / 32-K step.
__global__ __launch_bounds__(128) void qkv_kernel(
    const _Float16* __restrict__ xq, const _Float16* __restrict__ xc,
    const _Float16* __restrict__ wqT, const _Float16* __restrict__ wkT, const _Float16* __restrict__ wvT,
    const float* __restrict__ bq, const float* __restrict__ bk, const float* __restrict__ bv,
    const float* __restrict__ qn, const float* __restrict__ kn,
    _Float16* __restrict__ Qb, _Float16* __restrict__ Kb, _Float16* __restrict__ vTb)
{
  const int which = blockIdx.z;
  const int lane = threadIdx.x & 31;
  const int wave = threadIdx.x >> 5;
  const int ml = lane & 15;
  const int koff = (lane >> 4) * 8;
  const int m0 = blockIdx.x * 128 + wave * 32;
  const int n0 = blockIdx.y * 64;

  const _Float16* src = (which == 0) ? xq : xc;
  const _Float16* wT  = (which == 0) ? wqT : (which == 1 ? wkT : wvT);
  const float* bias   = (which == 0) ? bq  : (which == 1 ? bk  : bv);

  const _Float16* aptr0 = src + (size_t)src_row(m0 + ml) * Dv + koff;
  const _Float16* aptr1 = src + (size_t)src_row(m0 + 16 + ml) * Dv + koff;
  const _Float16* bptr[4];
#pragma unroll
  for (int n = 0; n < 4; n++) bptr[n] = wT + (size_t)(n0 + n * 16 + ml) * Dv + koff;

  v8f acc[2][4] = {};
  for (int k0 = 0; k0 < Dv; k0 += 32) {
    __builtin_prefetch(aptr0 + k0 + 64, 0, 0);
    __builtin_prefetch(aptr1 + k0 + 64, 0, 0);
    v16h a0 = mkfrag(ld8(aptr0 + k0), ld8(aptr0 + k0 + 16));
    v16h a1 = mkfrag(ld8(aptr1 + k0), ld8(aptr1 + k0 + 16));
#pragma unroll
    for (int n = 0; n < 4; n++) {
      v16h b = mkfrag(ld8(bptr[n] + k0), ld8(bptr[n] + k0 + 16));
      acc[0][n] = WMMA_F32_F16(a0, b, acc[0][n]);
      acc[1][n] = WMMA_F32_F16(a1, b, acc[1][n]);
    }
  }

  const int rowoff = (lane >> 4) * 8;
#pragma unroll
  for (int part = 0; part < 2; part++) {
    const int mp = m0 + part * 16;
    float val[4][8];
#pragma unroll
    for (int n = 0; n < 4; n++) {
      float bc = bias[n0 + n * 16 + ml];
#pragma unroll
      for (int i = 0; i < 8; i++) val[n][i] = acc[part][n][i] + bc;
    }

    if (which < 2) {                       // RMS-norm over the head dim (tile == one head)
      const float* nw_ = (which == 0) ? qn : kn;
      float ss[8];
#pragma unroll
      for (int i = 0; i < 8; i++) {
        ss[i] = 0.f;
#pragma unroll
        for (int n = 0; n < 4; n++) ss[i] += val[n][i] * val[n][i];
      }
#pragma unroll
      for (int msk = 1; msk < 16; msk <<= 1)
#pragma unroll
        for (int i = 0; i < 8; i++) ss[i] += __shfl_xor(ss[i], msk, 32);
      const float sc = (which == 0) ? 0.125f : 1.0f;   // fold HD^-0.5 into Q
      float rs[8];
#pragma unroll
      for (int i = 0; i < 8; i++) rs[i] = rsqrtf(ss[i] * (1.0f / 64.0f) + 1e-6f) * sc;
#pragma unroll
      for (int n = 0; n < 4; n++) {
        float wc = nw_[(n0 + n * 16 + ml) & 63];
#pragma unroll
        for (int i = 0; i < 8; i++) val[n][i] *= rs[i] * wc;
      }
    }

    if (which < 2) {
      _Float16* dst = (which == 0) ? Qb : Kb;
#pragma unroll
      for (int n = 0; n < 4; n++) {
        int col = n0 + n * 16 + ml;
#pragma unroll
        for (int i = 0; i < 8; i++)
          dst[(size_t)(mp + rowoff + i) * Dv + col] = (_Float16)val[n][i];
      }
    } else {                               // V transposed: [(w*H+h)*HD+d][token]
#pragma unroll
      for (int n = 0; n < 4; n++) {
        int col = n0 + n * 16 + ml;
        int hh = col >> 6, d = col & 63;
#pragma unroll
        for (int i = 0; i < 8; i++) {
          int r = mp + rowoff + i;
          int w = r >> 9, s = r & 511;
          vTb[((size_t)((w * Hv + hh) * HDv + d)) * Sv + s] = (_Float16)val[n][i];
        }
      }
    }
  }
}

// ---------------------------------------------------------------- flash attention
// grid (8 q-chunks, 16 heads, 8 windows), block = 128 (4 waves, 16 q-rows each)
__global__ __launch_bounds__(128) void attn_kernel(
    const _Float16* __restrict__ Qb, const _Float16* __restrict__ Kb,
    const _Float16* __restrict__ vTb, _Float16* __restrict__ aout,
    const float* __restrict__ rpbj)
{
  __shared__ __align__(16) _Float16 plds[4 * 16 * 32];
  const int lane = threadIdx.x & 31;
  const int wave = threadIdx.x >> 5;
  const int ml = lane & 15;
  const int koff = (lane >> 4) * 8;
  const int rowoff = (lane >> 4) * 8;
  const int qb = blockIdx.x, h = blockIdx.y, w = blockIdx.z;
  const int tq = qb;                      // 64-row chunks align to L

  const int qrow = qb * 64 + wave * 16;
  const _Float16* qp = Qb + (size_t)(w * Sv + qrow + ml) * Dv + h * HDv + koff;
  v16h qa0 = mkfrag(ld8(qp),      ld8(qp + 16));
  v16h qa1 = mkfrag(ld8(qp + 32), ld8(qp + 48));

  v8f accO[4] = {};
  float m_[8], l_[8];
#pragma unroll
  for (int i = 0; i < 8; i++) { m_[i] = -3.0e38f; l_[i] = 0.f; }

  _Float16* pw = plds + wave * 512;

  for (int ch = 0; ch < 16; ch++) {
    const int c0 = ch * 32;
    const int tk = c0 >> 6;
    const float biasv = rpbj[(tq - tk + WSv - 1) * Hv + h];

    v8f s0 = {}, s1 = {};
    {
      const _Float16* kp0 = Kb + (size_t)(w * Sv + c0 + ml) * Dv + h * HDv;
      const _Float16* kp1 = Kb + (size_t)(w * Sv + c0 + 16 + ml) * Dv + h * HDv;
      v16h kb00 = mkfrag(ld8(kp0 + koff),      ld8(kp0 + koff + 16));
      v16h kb01 = mkfrag(ld8(kp0 + 32 + koff), ld8(kp0 + 48 + koff));
      v16h kb10 = mkfrag(ld8(kp1 + koff),      ld8(kp1 + koff + 16));
      v16h kb11 = mkfrag(ld8(kp1 + 32 + koff), ld8(kp1 + 48 + koff));
      s0 = WMMA_F32_F16(qa0, kb00, s0);
      s0 = WMMA_F32_F16(qa1, kb01, s0);
      s1 = WMMA_F32_F16(qa0, kb10, s1);
      s1 = WMMA_F32_F16(qa1, kb11, s1);
    }
#pragma unroll
    for (int i = 0; i < 8; i++) { s0[i] += biasv; s1[i] += biasv; }

    float mn[8];
#pragma unroll
    for (int i = 0; i < 8; i++) mn[i] = fmaxf(s0[i], s1[i]);
#pragma unroll
    for (int msk = 1; msk < 16; msk <<= 1)
#pragma unroll
      for (int i = 0; i < 8; i++) mn[i] = fmaxf(mn[i], __shfl_xor(mn[i], msk, 32));

    float alpha[8], p0[8], p1[8], rsum[8];
#pragma unroll
    for (int i = 0; i < 8; i++) {
      mn[i] = fmaxf(mn[i], m_[i]);
      alpha[i] = __expf(m_[i] - mn[i]);
      m_[i] = mn[i];
      p0[i] = __expf(s0[i] - mn[i]);
      p1[i] = __expf(s1[i] - mn[i]);
      rsum[i] = p0[i] + p1[i];
    }
#pragma unroll
    for (int msk = 1; msk < 16; msk <<= 1)
#pragma unroll
      for (int i = 0; i < 8; i++) rsum[i] += __shfl_xor(rsum[i], msk, 32);
#pragma unroll
    for (int i = 0; i < 8; i++) l_[i] = l_[i] * alpha[i] + rsum[i];
#pragma unroll
    for (int n = 0; n < 4; n++)
#pragma unroll
      for (int i = 0; i < 8; i++) accO[n][i] *= alpha[i];

    // C-layout -> A-layout transpose of P through per-wave LDS
#pragma unroll
    for (int i = 0; i < 8; i++) {
      int rt = rowoff + i;
      pw[rt * 32 + ml]      = (_Float16)p0[i];
      pw[rt * 32 + ml + 16] = (_Float16)p1[i];
    }
    asm volatile("s_wait_dscnt 0" ::: "memory");
    const _Float16* pl = pw + ml * 32 + koff;
    v16h pa = mkfrag(ld8(pl), ld8(pl + 16));

#pragma unroll
    for (int n = 0; n < 4; n++) {
      const _Float16* vp = vTb + ((size_t)((w * Hv + h) * HDv + n * 16 + ml)) * Sv + c0 + koff;
      v16h vb = mkfrag(ld8(vp), ld8(vp + 16));
      accO[n] = WMMA_F32_F16(pa, vb, accO[n]);
    }
  }

  float inv[8];
#pragma unroll
  for (int i = 0; i < 8; i++) inv[i] = 1.0f / l_[i];
#pragma unroll
  for (int n = 0; n < 4; n++) {
    int col = h * HDv + n * 16 + ml;
#pragma unroll
    for (int i = 0; i < 8; i++) {
      int r = w * Sv + qrow + rowoff + i;
      aout[(size_t)r * Dv + col] = (_Float16)(accO[n][i] * inv[i]);
    }
  }
}

// ---------------------------------------------------------------- output projection + residual + roll(+SH)
// grid (32, 16): 128x64 tile per block, 32 rows per wave.
__global__ __launch_bounds__(128) void proj_kernel(
    const _Float16* __restrict__ aoutb, const _Float16* __restrict__ wpT,
    const float* __restrict__ bp, const float* __restrict__ qmf,
    float* __restrict__ feat, _Float16* __restrict__ cat16, int j)
{
  const int lane = threadIdx.x & 31;
  const int wave = threadIdx.x >> 5;
  const int ml = lane & 15;
  const int koff = (lane >> 4) * 8;
  const int m0 = blockIdx.x * 128 + wave * 32;
  const int n0 = blockIdx.y * 64;

  const _Float16* aptr0 = aoutb + (size_t)(m0 + ml) * Dv + koff;
  const _Float16* aptr1 = aoutb + (size_t)(m0 + 16 + ml) * Dv + koff;
  const _Float16* bptr[4];
#pragma unroll
  for (int n = 0; n < 4; n++) bptr[n] = wpT + (size_t)(n0 + n * 16 + ml) * Dv + koff;

  v8f acc[2][4] = {};
  for (int k0 = 0; k0 < Dv; k0 += 32) {
    __builtin_prefetch(aptr0 + k0 + 64, 0, 0);
    __builtin_prefetch(aptr1 + k0 + 64, 0, 0);
    v16h a0 = mkfrag(ld8(aptr0 + k0), ld8(aptr0 + k0 + 16));
    v16h a1 = mkfrag(ld8(aptr1 + k0), ld8(aptr1 + k0 + 16));
#pragma unroll
    for (int n = 0; n < 4; n++) {
      v16h b = mkfrag(ld8(bptr[n] + k0), ld8(bptr[n] + k0 + 16));
      acc[0][n] = WMMA_F32_F16(a0, b, acc[0][n]);
      acc[1][n] = WMMA_F32_F16(a1, b, acc[1][n]);
    }
  }

  const int rowoff = (lane >> 4) * 8;
#pragma unroll
  for (int part = 0; part < 2; part++) {
    const int mp = m0 + part * 16;
    int featrow[8];
#pragma unroll
    for (int i = 0; i < 8; i++) {
      int r = mp + rowoff + i;
      int w = r >> 9, s = r & 511;
      int b = w >> 2, widx = w & 3;
      int tqq = s >> 6, l = s & 63;
      int t = (widx * WSv + tqq + SHv) & 31;   // roll back by +SH
      featrow[i] = (b * Tv + t) * Lv + l;
    }
#pragma unroll
    for (int n = 0; n < 4; n++) {
      int col = n0 + n * 16 + ml;
      float bc = bp[col];
#pragma unroll
      for (int i = 0; i < 8; i++) {
        float v = acc[part][n][i] + bc + qmf[(size_t)featrow[i] * Dv + col];
        feat[(size_t)featrow[i] * Dv + col] = v;
        cat16[(size_t)featrow[i] * D3 + j * Dv + col] = (_Float16)v;
      }
    }
  }
}

// ---------------------------------------------------------------- gating MLP layer 1 (+GELU)
// grid (32, 32): 128x64 tile per block.
__global__ __launch_bounds__(128) void gate1_kernel(
    const _Float16* __restrict__ cat16, const _Float16* __restrict__ wg1T,
    const float* __restrict__ bg1, _Float16* __restrict__ g116)
{
  const int lane = threadIdx.x & 31;
  const int wave = threadIdx.x >> 5;
  const int ml = lane & 15;
  const int koff = (lane >> 4) * 8;
  const int m0 = blockIdx.x * 128 + wave * 32;
  const int n0 = blockIdx.y * 64;

  const _Float16* aptr0 = cat16 + (size_t)(m0 + ml) * D3 + koff;
  const _Float16* aptr1 = cat16 + (size_t)(m0 + 16 + ml) * D3 + koff;
  const _Float16* bptr[4];
#pragma unroll
  for (int n = 0; n < 4; n++) bptr[n] = wg1T + (size_t)(n0 + n * 16 + ml) * D3 + koff;

  v8f acc[2][4] = {};
  for (int k0 = 0; k0 < D3; k0 += 32) {
    __builtin_prefetch(aptr0 + k0 + 64, 0, 0);
    __builtin_prefetch(aptr1 + k0 + 64, 0, 0);
    v16h a0 = mkfrag(ld8(aptr0 + k0), ld8(aptr0 + k0 + 16));
    v16h a1 = mkfrag(ld8(aptr1 + k0), ld8(aptr1 + k0 + 16));
#pragma unroll
    for (int n = 0; n < 4; n++) {
      v16h b = mkfrag(ld8(bptr[n] + k0), ld8(bptr[n] + k0 + 16));
      acc[0][n] = WMMA_F32_F16(a0, b, acc[0][n]);
      acc[1][n] = WMMA_F32_F16(a1, b, acc[1][n]);
    }
  }

  const int rowoff = (lane >> 4) * 8;
#pragma unroll
  for (int part = 0; part < 2; part++) {
    const int mp = m0 + part * 16;
#pragma unroll
    for (int n = 0; n < 4; n++) {
      int col = n0 + n * 16 + ml;
      float bc = bg1[col];
#pragma unroll
      for (int i = 0; i < 8; i++) {
        float x = acc[part][n][i] + bc;
        float g = 0.5f * x * (1.0f + erff(x * 0.70710678118f));   // exact GELU
        g116[(size_t)(mp + rowoff + i) * D2 + col] = (_Float16)g;
      }
    }
  }
}

// ---------------------------------------------------------------- gating layer 2 (N=3) + softmax
__global__ __launch_bounds__(256) void gate2_kernel(
    const _Float16* __restrict__ g116, const float* __restrict__ Wg2,
    const float* __restrict__ bg2, float* __restrict__ gates)
{
  int row = blockIdx.x * 8 + (threadIdx.x >> 5);
  int lane = threadIdx.x & 31;
  if (row >= ROWS) return;
  const _Float16* gr = g116 + (size_t)row * D2;
  float a0 = 0.f, a1 = 0.f, a2 = 0.f;
  for (int k = lane; k < D2; k += 32) {
    float g = (float)gr[k];
    a0 = fmaf(g, Wg2[k * 3 + 0], a0);
    a1 = fmaf(g, Wg2[k * 3 + 1], a1);
    a2 = fmaf(g, Wg2[k * 3 + 2], a2);
  }
#pragma unroll
  for (int msk = 16; msk >= 1; msk >>= 1) {
    a0 += __shfl_xor(a0, msk, 32);
    a1 += __shfl_xor(a1, msk, 32);
    a2 += __shfl_xor(a2, msk, 32);
  }
  if (lane == 0) {
    a0 += bg2[0]; a1 += bg2[1]; a2 += bg2[2];
    float mx = fmaxf(a0, fmaxf(a1, a2));
    float e0 = __expf(a0 - mx), e1 = __expf(a1 - mx), e2 = __expf(a2 - mx);
    float inv = 1.0f / (e0 + e1 + e2);
    gates[row * 3 + 0] = e0 * inv;
    gates[row * 3 + 1] = e1 * inv;
    gates[row * 3 + 2] = e2 * inv;
  }
}

// ---------------------------------------------------------------- fuse
__global__ void fuse_kernel(const float* __restrict__ feats, const float* __restrict__ gates,
                            float* __restrict__ out)
{
  size_t i = (size_t)blockIdx.x * 256 + threadIdx.x;
  int row = (int)(i >> 10);
  const size_t NE = (size_t)ROWS * Dv;
  out[i] = gates[row * 3 + 0] * feats[i]
         + gates[row * 3 + 1] * feats[NE + i]
         + gates[row * 3 + 2] * feats[2 * NE + i];
}

// ---------------------------------------------------------------- launch
extern "C" void kernel_launch(void* const* d_in, const int* in_sizes, int n_in,
                              void* d_out, int out_size, void* d_ws, size_t ws_size,
                              hipStream_t stream)
{
  const float* audio = (const float*)d_in[0];
  const float* video = (const float*)d_in[1];
  const float* image = (const float*)d_in[2];
  const float* Wq  = (const float*)d_in[3];
  const float* Wk  = (const float*)d_in[4];
  const float* Wv  = (const float*)d_in[5];
  const float* Wp  = (const float*)d_in[6];
  const float* bq  = (const float*)d_in[7];
  const float* bk  = (const float*)d_in[8];
  const float* bv  = (const float*)d_in[9];
  const float* bp  = (const float*)d_in[10];
  const float* qn  = (const float*)d_in[11];
  const float* kn  = (const float*)d_in[12];
  const float* rpb = (const float*)d_in[13];
  const float* Wg1 = (const float*)d_in[14];
  const float* bg1 = (const float*)d_in[15];
  const float* Wg2 = (const float*)d_in[16];
  const float* bg2 = (const float*)d_in[17];
  float* out = (float*)d_out;

  char* p = (char*)d_ws;
  auto carve = [&](size_t bytes) -> void* {
    void* r = (void*)p;
    p += (bytes + 255) & ~(size_t)255;
    return r;
  };
  const size_t NE = (size_t)ROWS * Dv;                 // 4M elems
  _Float16* x16   = (_Float16*)carve(3 * NE * 2);
  _Float16* wqT   = (_Float16*)carve(3 * (size_t)Dv * Dv * 2);
  _Float16* wkT   = (_Float16*)carve(3 * (size_t)Dv * Dv * 2);
  _Float16* wvT   = (_Float16*)carve(3 * (size_t)Dv * Dv * 2);
  _Float16* wpT   = (_Float16*)carve(3 * (size_t)Dv * Dv * 2);
  _Float16* wg1T  = (_Float16*)carve((size_t)D3 * D2 * 2);
  _Float16* Qb    = (_Float16*)carve(NE * 2);
  _Float16* Kb    = (_Float16*)carve(NE * 2);
  _Float16* vTb   = (_Float16*)carve(NE * 2);
  _Float16* aoutb = (_Float16*)carve(NE * 2);
  _Float16* cat16 = (_Float16*)carve((size_t)ROWS * D3 * 2);
  _Float16* g116  = (_Float16*)carve((size_t)ROWS * D2 * 2);
  float*    feats = (float*)carve(3 * NE * 4);
  float*    gates = (float*)carve((size_t)ROWS * 3 * 4);

  // convert activations and transpose-convert weights to f16
  cvt_kernel<<<(int)(NE / 256), 256, 0, stream>>>(audio, x16 + 0 * NE, (int)NE);
  cvt_kernel<<<(int)(NE / 256), 256, 0, stream>>>(video, x16 + 1 * NE, (int)NE);
  cvt_kernel<<<(int)(NE / 256), 256, 0, stream>>>(image, x16 + 2 * NE, (int)NE);
  for (int j = 0; j < 3; j++) {
    size_t off = (size_t)j * Dv * Dv;
    cvtT_kernel<<<(Dv * Dv) / 256, 256, 0, stream>>>(Wq + off, wqT + off, Dv, Dv);
    cvtT_kernel<<<(Dv * Dv) / 256, 256, 0, stream>>>(Wk + off, wkT + off, Dv, Dv);
    cvtT_kernel<<<(Dv * Dv) / 256, 256, 0, stream>>>(Wv + off, wvT + off, Dv, Dv);
    cvtT_kernel<<<(Dv * Dv) / 256, 256, 0, stream>>>(Wp + off, wpT + off, Dv, Dv);
  }
  cvtT_kernel<<<(D3 * D2) / 256, 256, 0, stream>>>(Wg1, wg1T, D3, D2);

  const int qidx[3] = {0, 1, 2};
  const int cidx[3] = {1, 0, 1};
  const float* qsrcf[3] = {audio, video, image};
  for (int j = 0; j < 3; j++) {
    size_t woff = (size_t)j * Dv * Dv;
    qkv_kernel<<<dim3(32, 16, 3), 128, 0, stream>>>(
        x16 + (size_t)qidx[j] * NE, x16 + (size_t)cidx[j] * NE,
        wqT + woff, wkT + woff, wvT + woff,
        bq + j * Dv, bk + j * Dv, bv + j * Dv,
        qn + j * HDv, kn + j * HDv, Qb, Kb, vTb);
    attn_kernel<<<dim3(8, 16, 8), 128, 0, stream>>>(
        Qb, Kb, vTb, aoutb, rpb + j * (2 * WSv - 1) * Hv);
    proj_kernel<<<dim3(32, 16), 128, 0, stream>>>(
        aoutb, wpT + woff, bp + j * Dv, qsrcf[j],
        feats + (size_t)j * NE, cat16, j);
  }

  gate1_kernel<<<dim3(32, 32), 128, 0, stream>>>(cat16, wg1T, bg1, g116);
  gate2_kernel<<<ROWS / 8, 256, 0, stream>>>(g116, Wg2, bg2, gates);
  fuse_kernel<<<(int)(NE / 256), 256, 0, stream>>>(feats, gates, out);
}
